// TextGenerationModel_49237505081917
// MI455X (gfx1250) — matrix operand order repaired and checked by reading, not compiled, not run on hardware
//
#include <hip/hip_runtime.h>
#include <hip/hip_bf16.h>

// ---------------------------------------------------------------------------
// Types
// ---------------------------------------------------------------------------
typedef __bf16 bf16;
typedef __attribute__((ext_vector_type(16))) __bf16 v16bf;
typedef __attribute__((ext_vector_type(8)))  __bf16 v8bf;
typedef __attribute__((ext_vector_type(8)))  float  v8f;
typedef __attribute__((ext_vector_type(4)))  float  v4f;

#define VOCAB 10000
#define HID   256
#define BATCH 64
#define SEQ   256
#define G4H   1024   // 4*H

union U16bf { v16bf v; v8bf h[2]; };
union U8f   { v8f   v; v4f  f[2]; };

__device__ __forceinline__ v8f wmma_bf16(v16bf a, v16bf b, v8f c) {
  // v_wmma_f32_16x16x32_bf16  (D = A x B + C), wave32
  return __builtin_amdgcn_wmma_f32_16x16x32_bf16(false, a, false, b, (short)0, c,
                                                 false, false);
}

__device__ __forceinline__ v8f sigmoid8(v8f x) {
  v8f r;
#pragma unroll
  for (int i = 0; i < 8; ++i) r[i] = 1.0f / (1.0f + __expf(-x[i]));
  return r;
}
__device__ __forceinline__ v8f tanh8(v8f x) {
  v8f r;
#pragma unroll
  for (int i = 0; i < 8; ++i) r[i] = tanhf(x[i]);
  return r;
}

// gate-permutation: new row n' -> original row of a [4H, *] matrix.
// tile t' = n'/16 encodes (hidden-col-tile j = t'/4, gate q = t'%4).
__device__ __forceinline__ int perm_orig_row(int np) {
  int nt = np >> 4, j16 = np & 15;
  int q = nt & 3, jt = nt >> 2;
  return q * HID + jt * 16 + j16;
}

// ---------------------------------------------------------------------------
// Weight conversion: f32 [rows,256] -> bf16 [rows,256], optional gate permute
// ---------------------------------------------------------------------------
__global__ __launch_bounds__(256) void convert_w_kernel(const float* __restrict__ src,
                                                        bf16* __restrict__ dst,
                                                        int rows, int permute) {
  int idx = blockIdx.x * 256 + threadIdx.x;
  if (idx >= rows * HID) return;
  int r = idx >> 8, k = idx & 255;
  int orig = permute ? perm_orig_row(r) : r;
  dst[r * HID + k] = (bf16)src[orig * HID + k];
}

// combined bias with gate permute
__global__ __launch_bounds__(256) void bias_kernel(const float* __restrict__ bih,
                                                   const float* __restrict__ bhh,
                                                   float* __restrict__ bsum) {
  int idx = blockIdx.x * 256 + threadIdx.x;
  if (idx >= G4H) return;
  int orig = perm_orig_row(idx);
  bsum[idx] = bih[orig] + bhh[orig];
}

// ---------------------------------------------------------------------------
// Embedding gather:  xp0[t][b][n'] = W_ih0[orig(n')][x[b][t]] + bias0[n']
// stored C-tile swizzled: idx = (((t*64 + nt)*4 + m)*32 + lane)*8 + r
//   where b = m*16 + 8*(lane/16) + r,  col n' = nt*16 + lane%16
// Bias is folded here so the recurrent kernel does no bias math.
// ---------------------------------------------------------------------------
__global__ __launch_bounds__(256) void gather_xp0_kernel(const int* __restrict__ x,
                                                         const float* __restrict__ Wih0,
                                                         const float* __restrict__ bias,
                                                         float* __restrict__ xp0) {
  int idx = blockIdx.x * 256 + threadIdx.x;       // over SEQ*BATCH*G4H = 16.78M
  int t   = idx >> 16;
  int rem = idx & 65535;
  int b   = rem >> 10;
  int np  = rem & 1023;
  int nt = np >> 4, j16 = np & 15;
  int orig = perm_orig_row(np);
  int tok = x[b * SEQ + t];
  float val = Wih0[(size_t)orig * VOCAB + tok] + bias[np];
  int m = b >> 4;
  int l = (((b >> 3) & 1) << 4) + j16;
  int r = b & 7;
  xp0[(((size_t)(t * 64 + nt) * 4 + m) * 32 + l) * 8 + r] = val;
}

// ---------------------------------------------------------------------------
// Persistent LSTM layer: 1 workgroup, 512 threads = 16 waves, wave w owns
// hidden-col-tile w (16 cols) and its 4 gate N-tiles 4w..4w+3.
// h (bf16, [64][256]) lives in LDS; W_hh streamed bf16 from L2 each step.
// An opaque per-step integer offset defeats LICM on the weight loads while
// preserving global address-space inference (global_load, LOADcnt only).
// ---------------------------------------------------------------------------
__global__ __launch_bounds__(512, 1) void lstm_layer_kernel(
    const float* __restrict__ xp,     // swizzled [T][64nt][4m][32l][8r] f32 (+bias)
    const bf16*  __restrict__ Whh,    // [1024][256] bf16, gate-permuted rows
    bf16* __restrict__ hout)          // [T*64][256] bf16 row-major
{
  __shared__ bf16 hbuf[BATCH][HID];   // 32 KB
  const int tid = threadIdx.x;
  const int w = tid >> 5;             // wave 0..15 -> hidden tile
  const int l = tid & 31;
  const int l16 = l & 15;
  const int lhi = l >> 4;             // 0/1

  for (int i = tid; i < BATCH * HID; i += 512) ((bf16*)hbuf)[i] = (bf16)0.0f;

  v8f c[4];
#pragma unroll
  for (int m = 0; m < 4; ++m)
#pragma unroll
    for (int r = 0; r < 8; ++r) c[m][r] = 0.0f;

  // single base pointers; all per-tile deltas are immediate offsets
  const float* xp_w = xp + (size_t)w * 4096 + l * 8;       // += 65536 per step
  const bf16*  wh_w = Whh + (size_t)w * 16384 + l16 * 256 + lhi * 16;
  const bf16*  hb_r = &hbuf[l16][lhi * 16];                // LDS read base
  bf16*        hb_w = &hbuf[8 * lhi][16 * w + l16];        // LDS write base
  const uint4* cp_s = (const uint4*)hbuf + tid * 4;        // copy src (4x16B)
  uint4*       cp_d = (uint4*)hout + tid * 4;              // += 2048 per step

  __syncthreads();

  for (int t = 0; t < SEQ; ++t) {
    // ---- stream h_{t-1} (complete in LDS) out to global as b128s --------
    if (t > 0) {
#pragma unroll
      for (int i = 0; i < 4; ++i) cp_d[i] = cp_s[i];
      cp_d += 2048;
    }

    // ---- init accumulators = x-projection (bias pre-folded) -------------
    v8f acc[4][4];
#pragma unroll
    for (int q = 0; q < 4; ++q)
#pragma unroll
      for (int m = 0; m < 4; ++m) {
        const float* p = xp_w + (q * 4 + m) * 256;
        U8f u;
        u.f[0] = *(const v4f*)p;
        u.f[1] = *(const v4f*)(p + 4);
        acc[m][q] = u.v;
      }

    // ---- prefetch next step's x-projection tiles (global_prefetch) ------
    if (t + 1 < SEQ) {
#pragma unroll
      for (int q = 0; q < 4; ++q)
#pragma unroll
        for (int m = 0; m < 4; ++m)
          __builtin_prefetch(xp_w + 65536 + (q * 4 + m) * 256, 0, 3);
    }

    // Opaque zero offset: weight loads can't be hoisted out of the t-loop
    // (no spills), but the base keeps global address-space (global_load).
    unsigned woff = 0;
    asm volatile("" : "+v"(woff));
    const bf16* whp = wh_w + woff;

    // ---- gates += h_{t-1} @ Whh^T   (K = 256 in 8 chunks of 32) ---------
    // q processed in pairs to cap live B-operand registers.
#pragma unroll
    for (int k = 0; k < 8; ++k) {
      v16bf A[4];
#pragma unroll
      for (int m = 0; m < 4; ++m) {
        U16bf ua;
        ua.h[0] = *(const v8bf*)(hb_r + m * 4096 + k * 32);
        ua.h[1] = *(const v8bf*)(hb_r + m * 4096 + k * 32 + 8);
        A[m] = ua.v;
      }
#pragma unroll
      for (int qh = 0; qh < 2; ++qh) {
        v16bf B0 = *(const v16bf*)(whp + (2 * qh) * 4096 + k * 32);
        v16bf B1 = *(const v16bf*)(whp + (2 * qh + 1) * 4096 + k * 32);
#pragma unroll
        for (int m = 0; m < 4; ++m) acc[m][2 * qh]     = wmma_bf16(A[m], B0, acc[m][2 * qh]);
#pragma unroll
        for (int m = 0; m < 4; ++m) acc[m][2 * qh + 1] = wmma_bf16(A[m], B1, acc[m][2 * qh + 1]);
      }
    }

    __syncthreads();   // all hbuf reads (GEMM + copy) drained

    // ---- elementwise LSTM cell update, write h_t to LDS -----------------
#pragma unroll
    for (int m = 0; m < 4; ++m) {
      v8f i_ = sigmoid8(acc[m][0]);
      v8f f_ = sigmoid8(acc[m][1]);
      v8f g_ = tanh8(acc[m][2]);
      v8f o_ = sigmoid8(acc[m][3]);
      c[m] = f_ * c[m] + i_ * g_;
      v8f h_ = o_ * tanh8(c[m]);
#pragma unroll
      for (int r = 0; r < 8; ++r) hb_w[(m * 16 + r) * 256] = (bf16)h_[r];
    }

    xp_w += 65536;
    __syncthreads();   // hbuf fully updated before next step's reads
  }

  // final step's h
#pragma unroll
  for (int i = 0; i < 4; ++i) cp_d[i] = cp_s[i];
}

// ---------------------------------------------------------------------------
// GEMM: xp1 = h1 @ W_ih1p^T + bias1   (M=16384 rows = t*64+b, N=1024, K=256)
// output written in the C-tile-swizzled xp layout, bias folded at store.
// Block = 256 thr (8 waves). Wave tile = 64(M) x 32(N). WG tile = 64 x 256.
// ---------------------------------------------------------------------------
__global__ __launch_bounds__(256) void gemm_xp1_kernel(const bf16* __restrict__ h1,
                                                       const bf16* __restrict__ Wih1,
                                                       const float* __restrict__ bias,
                                                       float* __restrict__ xp1) {
  const int w = threadIdx.x >> 5, l = threadIdx.x & 31;
  const int l16 = l & 15, lhi = l >> 4;
  const int mt0 = blockIdx.x * 4;              // global M-tile base (of 1024)
  const int nt0 = blockIdx.y * 16 + w * 2;     // global N-tile base (of 64)

  v8f acc[4][2];
#pragma unroll
  for (int m = 0; m < 4; ++m)
#pragma unroll
    for (int j = 0; j < 2; ++j)
#pragma unroll
      for (int r = 0; r < 8; ++r) acc[m][j][r] = 0.0f;

  const bf16* a_base = h1 + (size_t)(mt0 * 16 + l16) * HID + lhi * 16;
  const bf16* b_base = Wih1 + (size_t)(nt0 * 16 + l16) * HID + lhi * 16;

#pragma unroll
  for (int k = 0; k < 8; ++k) {
    v16bf A[4];
#pragma unroll
    for (int m = 0; m < 4; ++m) {
      U16bf ua;
      ua.h[0] = *(const v8bf*)(a_base + m * 4096 + k * 32);
      ua.h[1] = *(const v8bf*)(a_base + m * 4096 + k * 32 + 8);
      A[m] = ua.v;
    }
#pragma unroll
    for (int j = 0; j < 2; ++j) {
      v16bf B = *(const v16bf*)(b_base + j * 4096 + k * 32);
#pragma unroll
      for (int m = 0; m < 4; ++m) acc[m][j] = wmma_bf16(A[m], B, acc[m][j]);
    }
  }

#pragma unroll
  for (int m = 0; m < 4; ++m) {
    const int gm = mt0 + m;
    const int t = gm >> 2, ml = gm & 3;
#pragma unroll
    for (int j = 0; j < 2; ++j) {
      const float bj = bias[(nt0 + j) * 16 + l16];
      float* dp = xp1 + (((size_t)(t * 64 + (nt0 + j)) * 4 + ml) * 32 + l) * 8;
      U8f u; u.v = acc[m][j];
#pragma unroll
      for (int r = 0; r < 8; ++r) u.v[r] += bj;
      *(v4f*)dp = u.f[0];
      *(v4f*)(dp + 4) = u.f[1];
    }
  }
}

// ---------------------------------------------------------------------------
// Final projection: out[b][v][t] = h2[t][b][:] . W_lin[v][:] + b_lin[v]
// GEMM rows ordered m = b*T + t so stores are t-contiguous b128s.
// N = 10000 = 625 N-tiles, guarded uniformly per wave-tile.
// ---------------------------------------------------------------------------
__global__ __launch_bounds__(256) void gemm_out_kernel(const bf16* __restrict__ h2,
                                                       const bf16* __restrict__ Wlin,
                                                       const float* __restrict__ blin,
                                                       float* __restrict__ out) {
  const int w = threadIdx.x >> 5, l = threadIdx.x & 31;
  const int l16 = l & 15, lhi = l >> 4;
  const int mt0 = blockIdx.x * 4;            // of 1024 M-tiles
  const int nt0 = blockIdx.y * 16 + w * 2;   // of 625 N-tiles

  v8f acc[4][2];
#pragma unroll
  for (int m = 0; m < 4; ++m)
#pragma unroll
    for (int j = 0; j < 2; ++j)
#pragma unroll
      for (int r = 0; r < 8; ++r) acc[m][j][r] = 0.0f;

  const bf16* b_base = Wlin + (size_t)(nt0 * 16 + l16) * HID + lhi * 16;

#pragma unroll
  for (int k = 0; k < 8; ++k) {
    v16bf A[4];
#pragma unroll
    for (int m = 0; m < 4; ++m) {
      const int mrow = (mt0 + m) * 16 + l16;   // m = b*T + t
      const int b = mrow >> 8, tt = mrow & 255;
      const bf16* ap = h2 + (size_t)(tt * 64 + b) * HID + lhi * 16;
      U16bf ua;
      ua.h[0] = *(const v8bf*)(ap + k * 32);
      ua.h[1] = *(const v8bf*)(ap + k * 32 + 8);
      A[m] = ua.v;
    }
#pragma unroll
    for (int j = 0; j < 2; ++j) {
      if (nt0 + j < 625) {
        v16bf B = *(const v16bf*)(b_base + j * 4096 + k * 32);
#pragma unroll
        for (int m = 0; m < 4; ++m) acc[m][j] = wmma_bf16(A[m], B, acc[m][j]);
      }
    }
  }

#pragma unroll
  for (int m = 0; m < 4; ++m) {
    const int gm = mt0 + m;
    const int b = gm >> 4;
    const int tb = (gm & 15) * 16 + 8 * lhi;   // 8 consecutive t per half-group
#pragma unroll
    for (int j = 0; j < 2; ++j) {
      if (nt0 + j < 625) {
        const int v = (nt0 + j) * 16 + l16;
        const float bl = blin[v];
        U8f u; u.v = acc[m][j];
#pragma unroll
        for (int r = 0; r < 8; ++r) u.v[r] += bl;
        float* dp = out + ((size_t)b * VOCAB + v) * SEQ + tb;
        *(v4f*)dp = u.f[0];
        *(v4f*)(dp + 4) = u.f[1];
      }
    }
  }
}

// ---------------------------------------------------------------------------
// Host launcher
// ---------------------------------------------------------------------------
extern "C" void kernel_launch(void* const* d_in, const int* in_sizes, int n_in,
                              void* d_out, int out_size, void* d_ws, size_t ws_size,
                              hipStream_t stream) {
  (void)in_sizes; (void)n_in; (void)out_size; (void)ws_size;
  const int*   x    = (const int*)  d_in[0];
  const float* Wih0 = (const float*)d_in[1];
  const float* Whh0 = (const float*)d_in[2];
  const float* bih0 = (const float*)d_in[3];
  const float* bhh0 = (const float*)d_in[4];
  const float* Wih1 = (const float*)d_in[5];
  const float* Whh1 = (const float*)d_in[6];
  const float* bih1 = (const float*)d_in[7];
  const float* bhh1 = (const float*)d_in[8];
  const float* Wlin = (const float*)d_in[9];
  const float* blin = (const float*)d_in[10];
  float* out = (float*)d_out;

  // workspace carve-up (256B aligned)
  char* ws = (char*)d_ws;
  size_t off = 0;
  auto carve = [&](size_t bytes) -> char* {
    char* p = ws + off;
    off = (off + bytes + 255) & ~(size_t)255;
    return p;
  };
  const size_t XPN = (size_t)SEQ * BATCH * G4H;          // 16.78M
  float* xp0   = (float*)carve(XPN * 4);
  float* xp1   = (float*)carve(XPN * 4);
  bf16*  h1    = (bf16*)carve((size_t)SEQ * BATCH * HID * 2);
  bf16*  h2    = (bf16*)carve((size_t)SEQ * BATCH * HID * 2);
  bf16*  Whh0b = (bf16*)carve((size_t)G4H * HID * 2);
  bf16*  Whh1b = (bf16*)carve((size_t)G4H * HID * 2);
  bf16*  Wih1b = (bf16*)carve((size_t)G4H * HID * 2);
  bf16*  Wlinb = (bf16*)carve((size_t)VOCAB * HID * 2);
  float* bias0 = (float*)carve(G4H * 4);
  float* bias1 = (float*)carve(G4H * 4);

  // 1) weight conversion (bf16, gate-permuted where needed)
  convert_w_kernel<<<(G4H * HID + 255) / 256, 256, 0, stream>>>(Whh0, Whh0b, G4H, 1);
  convert_w_kernel<<<(G4H * HID + 255) / 256, 256, 0, stream>>>(Whh1, Whh1b, G4H, 1);
  convert_w_kernel<<<(G4H * HID + 255) / 256, 256, 0, stream>>>(Wih1, Wih1b, G4H, 1);
  convert_w_kernel<<<(VOCAB * HID + 255) / 256, 256, 0, stream>>>(Wlin, Wlinb, VOCAB, 0);
  bias_kernel<<<4, 256, 0, stream>>>(bih0, bhh0, bias0);
  bias_kernel<<<4, 256, 0, stream>>>(bih1, bhh1, bias1);

  // 2) embedding gather into swizzled xp0 (+bias0 folded)
  gather_xp0_kernel<<<(unsigned)(XPN / 256), 256, 0, stream>>>(x, Wih0, bias0, xp0);

  // 3) layer 0 recurrence (persistent, 1 WGP, 16 waves)
  lstm_layer_kernel<<<1, 512, 0, stream>>>(xp0, Whh0b, h1);

  // 4) xp1 = h1 @ W_ih1p^T + bias1 (parallel WMMA GEMM, swizzled output)
  gemm_xp1_kernel<<<dim3(256, 4), 256, 0, stream>>>(h1, Wih1b, bias1, xp1);

  // 5) layer 1 recurrence
  lstm_layer_kernel<<<1, 512, 0, stream>>>(xp1, Whh1b, h2);

  // 6) vocab projection + bias, [B][V][T] output with contiguous-t stores
  gemm_out_kernel<<<dim3(256, 40), 256, 0, stream>>>(h2, Wlinb, blin, out);
}
